// TransformerTower_50775103373702
// MI455X (gfx1250) — matrix operand order repaired and checked by reading, not compiled, not run on hardware
//
#include <hip/hip_runtime.h>
#include <hip/hip_bf16.h>

// ---------------------------------------------------------------------------
// Types for CDNA5 WMMA (wave32)
// ---------------------------------------------------------------------------
typedef __attribute__((ext_vector_type(16))) __bf16 v16bf;
typedef __attribute__((ext_vector_type(8)))  __bf16 v8bf;
typedef __attribute__((ext_vector_type(8)))  float  v8f;

#define NB_B 2
#define NB_S 1024
#define NB_DIM 1024
#define NB_H 16
#define NB_HD 64
#define NB_DFF 2048
#define NB_DEPTH 6

__device__ __forceinline__ __bf16 f2bf(float x) {
    union { float f; unsigned u; } a; a.f = x;
    unsigned r = (a.u + 0x7fffu + ((a.u >> 16) & 1u)) >> 16;   // RNE
    union { unsigned short s; __bf16 b; } o; o.s = (unsigned short)r;
    return o.b;
}

__device__ __forceinline__ float bf2f(__bf16 b) {
    union { __bf16 b; unsigned short s; } i; i.b = b;
    union { unsigned u; float f; } o; o.u = ((unsigned)i.s) << 16;
    return o.f;
}

__device__ __forceinline__ v8f wmma_bf16(v16bf a, v16bf b, v8f c) {
    return __builtin_amdgcn_wmma_f32_16x16x32_bf16(
        /*neg_a=*/false, a, /*neg_b=*/false, b,
        /*c_mod=*/(short)0, c, /*reuse_a=*/false, /*reuse_b=*/false);
}

// A-fragment (16x32 bf16): lane holds row M=lane%16,
// K = 2*koff + {0..15}, koff = (lane>=16)*8  -> two 16B loads
__device__ __forceinline__ v16bf load_afrag_bf(const __bf16* base, int koff) {
    v8bf lo = *(const v8bf*)(base + 2 * koff);
    v8bf hi = *(const v8bf*)(base + 2 * koff + 8);
    v16bf f;
#pragma unroll
    for (int i = 0; i < 8; ++i) { f[i] = lo[i]; f[8 + i] = hi[i]; }
    return f;
}

// B-fragment (32x16 bf16, column-major): lane holds column N=lane%16,
// K = (lane>=16)*16 + {0..15}  -> one contiguous 32B load
__device__ __forceinline__ v16bf load_bfrag_bf(const __bf16* base, int hf) {
    return *(const v16bf*)(base + hf * 16);
}

// ---------------------------------------------------------------------------
// Elementwise f32 -> bf16 (weight staging; one pass per tensor per layer)
// ---------------------------------------------------------------------------
__global__ __launch_bounds__(256) void cvt_bf16_kernel(const float* __restrict__ in,
                                                       __bf16* __restrict__ out,
                                                       int n8) {
    const int idx = blockIdx.x * 256 + threadIdx.x;
    if (idx >= n8) return;
    const float4* p = (const float4*)(in + (size_t)idx * 8);
    float4 a = p[0], b = p[1];
    v8bf o;
    o[0] = f2bf(a.x); o[1] = f2bf(a.y); o[2] = f2bf(a.z); o[3] = f2bf(a.w);
    o[4] = f2bf(b.x); o[5] = f2bf(b.y); o[6] = f2bf(b.z); o[7] = f2bf(b.w);
    *(v8bf*)(out + (size_t)idx * 8) = o;
}

// ---------------------------------------------------------------------------
// RMSNorm: out(bf16)[row,:] = x * w * rsqrt(mean(x^2) + eps)
// ---------------------------------------------------------------------------
__global__ __launch_bounds__(256) void rms_kernel(const float* __restrict__ h,
                                                  const float* __restrict__ w,
                                                  __bf16* __restrict__ out) {
    const int row = blockIdx.x;
    const float* x = h + (size_t)row * NB_DIM;
    float s = 0.f;
    for (int i = threadIdx.x; i < NB_DIM; i += 256) { float v = x[i]; s += v * v; }
    __shared__ float red[256];
    red[threadIdx.x] = s;
    __syncthreads();
    for (int off = 128; off > 0; off >>= 1) {
        if ((int)threadIdx.x < off) red[threadIdx.x] += red[threadIdx.x + off];
        __syncthreads();
    }
    const float inv = rsqrtf(red[0] * (1.f / (float)NB_DIM) + 1e-8f);
    __bf16* o = out + (size_t)row * NB_DIM;
    for (int i = threadIdx.x; i < NB_DIM; i += 256) o[i] = f2bf(x[i] * w[i] * inv);
}

// ---------------------------------------------------------------------------
// GEMM: C[M,N] = A[M,K](bf16) @ W[N,K](bf16)^T + bias[N] (+res f32)(opt relu)
// Writes f32 (Cf) and/or bf16 (Cb). Block = 256 thr (8 waves, 2x4),
// block tile 128x128, wave tile 64x32 (4x2 of 16x16 WMMA).
// 12 b128 loads -> 8 v_wmma per k-step; 1024 waves for the 2048x1024 GEMMs.
// ---------------------------------------------------------------------------
__global__ __launch_bounds__(256) void gemm_kernel(
    const __bf16* __restrict__ A, const __bf16* __restrict__ W,
    const float* __restrict__ bias, const float* __restrict__ res,
    float* __restrict__ Cf, __bf16* __restrict__ Cb,
    int M, int N, int K, int relu) {
    const int lane = threadIdx.x & 31;
    const int wave = threadIdx.x >> 5;
    const int r    = lane & 15;
    const int hf   = lane >> 4;
    const int koff = hf * 8;
    const int m_base = blockIdx.y * 128 + (wave & 1) * 64;
    const int n_base = blockIdx.x * 128 + (wave >> 1) * 32;

    v8f acc[4][2];
#pragma unroll
    for (int ms = 0; ms < 4; ++ms)
#pragma unroll
        for (int ns = 0; ns < 2; ++ns) { v8f z = {}; acc[ms][ns] = z; }

    for (int k0 = 0; k0 < K; k0 += 32) {
        v16bf af[4], bfm[2];
#pragma unroll
        for (int ms = 0; ms < 4; ++ms)
            af[ms] = load_afrag_bf(A + (size_t)(m_base + ms * 16 + r) * K + k0, koff);
#pragma unroll
        for (int ns = 0; ns < 2; ++ns)
            bfm[ns] = load_bfrag_bf(W + (size_t)(n_base + ns * 16 + r) * K + k0, hf);
#pragma unroll
        for (int ms = 0; ms < 4; ++ms)
#pragma unroll
            for (int ns = 0; ns < 2; ++ns)
                acc[ms][ns] = wmma_bf16(af[ms], bfm[ns], acc[ms][ns]);
    }

#pragma unroll
    for (int ms = 0; ms < 4; ++ms) {
#pragma unroll
        for (int ns = 0; ns < 2; ++ns) {
            const int n = n_base + ns * 16 + r;
            const float bn = bias[n];
#pragma unroll
            for (int i = 0; i < 8; ++i) {
                const int m = m_base + ms * 16 + i + hf * 8;
                float v = acc[ms][ns][i] + bn;
                if (relu) v = fmaxf(v, 0.f);
                if (res)  v += res[(size_t)m * N + n];
                if (Cf) Cf[(size_t)m * N + n] = v;
                if (Cb) Cb[(size_t)m * N + n] = f2bf(v);
            }
        }
    }
}

// ---------------------------------------------------------------------------
// RoPE + repack: q,k (bf16 [B,S,DIM]) -> bf16 [B*H, S, 64] rotary-applied;
// v -> bf16 transposed [B*H, 64, S] (so P@V's B-operand is j-contiguous).
// ---------------------------------------------------------------------------
__global__ __launch_bounds__(256) void rope_pack_kernel(
    const __bf16* __restrict__ q, const __bf16* __restrict__ k,
    const __bf16* __restrict__ v, __bf16* __restrict__ qb,
    __bf16* __restrict__ kb, __bf16* __restrict__ vt) {
    const int token = blockIdx.x;          // b*S + s
    const int b = token >> 10;
    const int s = token & (NB_S - 1);
    const size_t in_base = (size_t)token * NB_DIM;
    const float spos = (float)s;
    for (int t = threadIdx.x; t < NB_DIM; t += 256) {
        const int hh = t >> 6;
        const int d  = t & 63;
        const int i  = d & 31;             // freq index
        // freq_i = 2048^(-i/31) = exp(-i * ln(2048)/31)
        const float fr = __expf(-(float)i * 0.24595545116f);
        const float th = spos * fr;
        float sn, cs;
        __sincosf(th, &sn, &cs);
        const size_t qk_idx = (((size_t)(b * NB_H + hh)) * NB_S + s) * NB_HD + d;
        const float q1 = bf2f(q[in_base + hh * 64 + 2 * i]);
        const float q2 = bf2f(q[in_base + hh * 64 + 2 * i + 1]);
        const float k1 = bf2f(k[in_base + hh * 64 + 2 * i]);
        const float k2 = bf2f(k[in_base + hh * 64 + 2 * i + 1]);
        const float oq = (d < 32) ? (q1 * cs - q2 * sn) : (q1 * sn + q2 * cs);
        const float ok = (d < 32) ? (k1 * cs - k2 * sn) : (k1 * sn + k2 * cs);
        qb[qk_idx] = f2bf(oq);
        kb[qk_idx] = f2bf(ok);
        vt[((size_t)(b * NB_H + hh) * NB_HD + d) * NB_S + s] = v[in_base + hh * 64 + d];
    }
}

// ---------------------------------------------------------------------------
// Fused flash attention: 128-thread block = 4 waves, each wave owns one
// (b,h, 16-query tile). Online softmax; QK^T and P@V via bf16 WMMA.
// ctx written bf16 [B,S,DIM].
// ---------------------------------------------------------------------------
__global__ __launch_bounds__(128) void attn_kernel(
    const __bf16* __restrict__ qb, const __bf16* __restrict__ kb,
    const __bf16* __restrict__ vt, __bf16* __restrict__ ctx) {
    const int wave = threadIdx.x >> 5;
    const int gidx = blockIdx.x * 4 + wave;     // global tile index
    const int tile = gidx & 63;                 // S/16 = 64 query tiles
    const int bh   = gidx >> 6;
    const int hh   = bh & (NB_H - 1);
    const int b    = bh >> 4;
    const __bf16* q = qb + (size_t)bh * NB_S * NB_HD;
    const __bf16* k = kb + (size_t)bh * NB_S * NB_HD;
    const __bf16* v = vt + (size_t)bh * NB_HD * NB_S;
    const int lane = threadIdx.x & 31;
    const int r    = lane & 15;
    const int hf   = lane >> 4;
    const int koff = hf * 8;
    const int s0   = tile * 16;

    const v16bf qf0 = load_afrag_bf(q + (size_t)(s0 + r) * NB_HD, koff);       // d 0..31
    const v16bf qf1 = load_afrag_bf(q + (size_t)(s0 + r) * NB_HD + 32, koff);  // d 32..63

    v8f o0 = {}, o1 = {}, o2 = {}, o3 = {};
    float m_run = -1e30f, l_run = 0.f;     // per-row stats, rows live in lanes 0..15
    __shared__ float sP[4][16][32];        // per-wave score panel
    __shared__ float sRow[4][16];
    float (*P)[32] = sP[wave];
    float* R = sRow[wave];

    for (int j0 = 0; j0 < NB_S; j0 += 32) {
        v8f sc0 = {}, sc1 = {};
        {   // keys j0..j0+15
            const __bf16* kr = k + (size_t)(j0 + r) * NB_HD;
            sc0 = wmma_bf16(qf0, load_bfrag_bf(kr, hf), sc0);
            sc0 = wmma_bf16(qf1, load_bfrag_bf(kr + 32, hf), sc0);
        }
        {   // keys j0+16..j0+31
            const __bf16* kr = k + (size_t)(j0 + 16 + r) * NB_HD;
            sc1 = wmma_bf16(qf0, load_bfrag_bf(kr, hf), sc1);
            sc1 = wmma_bf16(qf1, load_bfrag_bf(kr + 32, hf), sc1);
        }
#pragma unroll
        for (int i = 0; i < 8; ++i) {      // C layout: row = i + 8*hf, col = r
            P[i + hf * 8][r]      = sc0[i] * 0.125f;   // 1/sqrt(64)
            P[i + hf * 8][16 + r] = sc1[i] * 0.125f;
        }
        __syncthreads();
        if (lane < 16) {                   // row = lane: online softmax update
            float mx = m_run;
#pragma unroll
            for (int j = 0; j < 32; ++j) mx = fmaxf(mx, P[lane][j]);
            const float corr = __expf(m_run - mx);
            float sum = 0.f;
#pragma unroll
            for (int j = 0; j < 32; ++j) {
                const float e = __expf(P[lane][j] - mx);
                P[lane][j] = e;
                sum += e;
            }
            l_run = l_run * corr + sum;
            m_run = mx;
            R[lane] = corr;
        }
        __syncthreads();
#pragma unroll
        for (int i = 0; i < 8; ++i) {      // rescale running O
            const float f = R[i + hf * 8];
            o0[i] *= f; o1[i] *= f; o2[i] *= f; o3[i] *= f;
        }
        v16bf pf;                          // P as A-fragment (16x32 over j)
#pragma unroll
        for (int t = 0; t < 8; ++t) {
            pf[t]     = f2bf(P[r][2 * koff + t]);
            pf[8 + t] = f2bf(P[r][2 * koff + 8 + t]);
        }
        o0 = wmma_bf16(pf, load_bfrag_bf(v + (size_t)(r)      * NB_S + j0, hf), o0);
        o1 = wmma_bf16(pf, load_bfrag_bf(v + (size_t)(16 + r) * NB_S + j0, hf), o1);
        o2 = wmma_bf16(pf, load_bfrag_bf(v + (size_t)(32 + r) * NB_S + j0, hf), o2);
        o3 = wmma_bf16(pf, load_bfrag_bf(v + (size_t)(48 + r) * NB_S + j0, hf), o3);
        __syncthreads();                   // sP reused next iteration
    }

    if (lane < 16) R[lane] = 1.f / l_run;
    __syncthreads();
#pragma unroll
    for (int i = 0; i < 8; ++i) {
        const float f = R[i + hf * 8];
        const size_t rowp =
            ((size_t)b * NB_S + s0 + i + hf * 8) * NB_DIM + hh * NB_HD;
        ctx[rowp + r]      = f2bf(o0[i] * f);
        ctx[rowp + 16 + r] = f2bf(o1[i] * f);
        ctx[rowp + 32 + r] = f2bf(o2[i] * f);
        ctx[rowp + 48 + r] = f2bf(o3[i] * f);
    }
}

// ---------------------------------------------------------------------------
// Host launcher
// ---------------------------------------------------------------------------
extern "C" void kernel_launch(void* const* d_in, const int* in_sizes, int n_in,
                              void* d_out, int out_size, void* d_ws, size_t ws_size,
                              hipStream_t stream) {
    (void)in_sizes; (void)n_in; (void)out_size; (void)ws_size;
    const float* x    = (const float*)d_in[0];
    const float* ln1w = (const float*)d_in[1];
    const float* wq   = (const float*)d_in[2];
    const float* bq   = (const float*)d_in[3];
    const float* wk   = (const float*)d_in[4];
    const float* bk   = (const float*)d_in[5];
    const float* wv   = (const float*)d_in[6];
    const float* bv   = (const float*)d_in[7];
    const float* wo   = (const float*)d_in[8];
    const float* bo   = (const float*)d_in[9];
    const float* ln2w = (const float*)d_in[10];
    const float* w1   = (const float*)d_in[11];
    const float* b1   = (const float*)d_in[12];
    const float* w2   = (const float*)d_in[13];
    const float* b2   = (const float*)d_in[14];

    float* h = (float*)d_out;              // residual stream lives in d_out (f32)

    const size_t NT = (size_t)NB_B * NB_S; // 2048 tokens
    char* wsp = (char*)d_ws;
    // per-layer bf16 weight staging
    __bf16* lw_q = (__bf16*)wsp; wsp += (size_t)NB_DIM * NB_DIM * 2;
    __bf16* lw_k = (__bf16*)wsp; wsp += (size_t)NB_DIM * NB_DIM * 2;
    __bf16* lw_v = (__bf16*)wsp; wsp += (size_t)NB_DIM * NB_DIM * 2;
    __bf16* lw_o = (__bf16*)wsp; wsp += (size_t)NB_DIM * NB_DIM * 2;
    __bf16* lw_1 = (__bf16*)wsp; wsp += (size_t)NB_DFF * NB_DIM * 2;
    __bf16* lw_2 = (__bf16*)wsp; wsp += (size_t)NB_DIM * NB_DFF * 2;
    // activations (bf16)
    __bf16* buf_hn = (__bf16*)wsp; wsp += NT * NB_DIM * 2;
    __bf16* buf_q  = (__bf16*)wsp; wsp += NT * NB_DIM * 2;
    __bf16* buf_k  = (__bf16*)wsp; wsp += NT * NB_DIM * 2;
    __bf16* buf_v  = (__bf16*)wsp; wsp += NT * NB_DIM * 2;
    __bf16* qhat   = (__bf16*)wsp; wsp += NT * NB_DIM * 2;
    __bf16* khat   = (__bf16*)wsp; wsp += NT * NB_DIM * 2;
    __bf16* vtb    = (__bf16*)wsp; wsp += NT * NB_DIM * 2;
    __bf16* buf_u  = (__bf16*)wsp; wsp += NT * NB_DFF * 2;
    __bf16* buf_cx = (__bf16*)wsp; wsp += NT * NB_DIM * 2;

    hipMemcpyAsync(h, x, NT * NB_DIM * sizeof(float),
                   hipMemcpyDeviceToDevice, stream);

    const int M = (int)NT;
    const dim3 gProj(NB_DIM / 128, M / 128);   // 8 x 16
    const dim3 gFF1(NB_DFF / 128, M / 128);    // 16 x 16
    const int n8_sq = NB_DIM * NB_DIM / 8;     // 131072
    const int n8_ff = NB_DFF * NB_DIM / 8;     // 262144

    for (int l = 0; l < NB_DEPTH; ++l) {
        const size_t wofs  = (size_t)l * NB_DIM * NB_DIM;
        const size_t wofs2 = (size_t)l * NB_DFF * NB_DIM;

        // stage this layer's weights as bf16
        cvt_bf16_kernel<<<n8_sq / 256, 256, 0, stream>>>(wq + wofs, lw_q, n8_sq);
        cvt_bf16_kernel<<<n8_sq / 256, 256, 0, stream>>>(wk + wofs, lw_k, n8_sq);
        cvt_bf16_kernel<<<n8_sq / 256, 256, 0, stream>>>(wv + wofs, lw_v, n8_sq);
        cvt_bf16_kernel<<<n8_sq / 256, 256, 0, stream>>>(wo + wofs, lw_o, n8_sq);
        cvt_bf16_kernel<<<n8_ff / 256, 256, 0, stream>>>(w1 + wofs2, lw_1, n8_ff);
        cvt_bf16_kernel<<<n8_ff / 256, 256, 0, stream>>>(w2 + wofs2, lw_2, n8_ff);

        rms_kernel<<<M, 256, 0, stream>>>(h, ln1w + (size_t)l * NB_DIM, buf_hn);

        gemm_kernel<<<gProj, 256, 0, stream>>>(buf_hn, lw_q, bq + (size_t)l * NB_DIM,
                                               nullptr, nullptr, buf_q,
                                               M, NB_DIM, NB_DIM, 0);
        gemm_kernel<<<gProj, 256, 0, stream>>>(buf_hn, lw_k, bk + (size_t)l * NB_DIM,
                                               nullptr, nullptr, buf_k,
                                               M, NB_DIM, NB_DIM, 0);
        gemm_kernel<<<gProj, 256, 0, stream>>>(buf_hn, lw_v, bv + (size_t)l * NB_DIM,
                                               nullptr, nullptr, buf_v,
                                               M, NB_DIM, NB_DIM, 0);

        rope_pack_kernel<<<M, 256, 0, stream>>>(buf_q, buf_k, buf_v, qhat, khat, vtb);

        attn_kernel<<<NB_B * NB_H * (NB_S / 16) / 4, 128, 0, stream>>>(qhat, khat, vtb, buf_cx);

        gemm_kernel<<<gProj, 256, 0, stream>>>(buf_cx, lw_o, bo + (size_t)l * NB_DIM,
                                               h, h, nullptr, M, NB_DIM, NB_DIM, 0);

        rms_kernel<<<M, 256, 0, stream>>>(h, ln2w + (size_t)l * NB_DIM, buf_hn);

        gemm_kernel<<<gFF1, 256, 0, stream>>>(buf_hn, lw_1, b1 + (size_t)l * NB_DFF,
                                              nullptr, nullptr, buf_u,
                                              M, NB_DFF, NB_DIM, 1);
        gemm_kernel<<<gProj, 256, 0, stream>>>(buf_u, lw_2, b2 + (size_t)l * NB_DIM,
                                               h, h, nullptr, M, NB_DIM, NB_DFF, 0);
    }
}
